// SinkhornStableNormalizer_64665027609306
// MI455X (gfx1250) — matrix operand order repaired and checked by reading, not compiled, not run on hardware
//
#include <hip/hip_runtime.h>

// Sinkhorn normalization, factored form:
//   M = exp(x) + eps ;  r <- 1/(M c) ; c <- 1/(M^T r)  (x100) ; y = diag(r) M diag(c)
// One workgroup (32 waves) per batch. GEMV via v_wmma_f32_16x16x32_bf16 with the
// vector broadcast across B's 16 columns; bf16 hi/lo splitting of both the matrix
// and the vector (3 WMMAs per K-tile) recovers ~f32 accuracy.
// Workspace layout (requires 32 MiB): Mhi | Mlo | MThi | MTlo, each 16*512*512 bf16.

typedef __attribute__((ext_vector_type(16))) __bf16 v16bf;
typedef __attribute__((ext_vector_type(8)))  __bf16 bf16x8;
typedef __attribute__((ext_vector_type(8)))  float  v8f;

#define B_ 16
#define N_ 512
#define EPS_ 0.001f
#define ITERS_ 100

// ---------------------------------------------------------------------------
// Prep: M = exp(x)+eps, split into bf16 hi/lo planes, plus transposed planes
// via an LDS 32x33 tile (conflict-free, coalesced on both sides).
// ---------------------------------------------------------------------------
__global__ __launch_bounds__(256) void sinkhorn_prep(
    const float* __restrict__ x,
    __bf16* __restrict__ Mhi, __bf16* __restrict__ Mlo,
    __bf16* __restrict__ MThi, __bf16* __restrict__ MTlo)
{
    __shared__ __attribute__((aligned(32))) __bf16 tH[32][33];
    __shared__ __attribute__((aligned(32))) __bf16 tL[32][33];

    const int b  = blockIdx.z;
    const int ti = blockIdx.y;   // row-tile
    const int tj = blockIdx.x;   // col-tile
    const size_t mb = (size_t)b * N_ * N_;

#pragma unroll
    for (int k = 0; k < 4; ++k) {
        int l  = threadIdx.x + k * 256;
        int li = l >> 5, lj = l & 31;
        int gi = ti * 32 + li, gj = tj * 32 + lj;
        float m = expf(x[mb + (size_t)gi * N_ + gj]) + EPS_;
        __bf16 h  = (__bf16)m;
        __bf16 lo = (__bf16)(m - (float)h);
        Mhi[mb + (size_t)gi * N_ + gj] = h;
        Mlo[mb + (size_t)gi * N_ + gj] = lo;
        tH[li][lj] = h;
        tL[li][lj] = lo;
    }
    __syncthreads();
#pragma unroll
    for (int k = 0; k < 4; ++k) {
        int l  = threadIdx.x + k * 256;
        int li = l >> 5, lj = l & 31;
        int gi = tj * 32 + li, gj = ti * 32 + lj;   // transposed tile coords
        MThi[mb + (size_t)gi * N_ + gj] = tH[lj][li];
        MTlo[mb + (size_t)gi * N_ + gj] = tL[lj][li];
    }
}

// ---------------------------------------------------------------------------
// GEMV for one workgroup: each of 32 waves owns one 16-row tile of Mh/Ml.
// B broadcasts the vector across all 16 columns => every D column = M*v.
// A layout (16-bit 16x32): lanes 0-15 row L, e0..7=K0..7, e8..15=K16..23;
// lanes 16-31 row L-16, e0..7=K8..15, e8..15=K24..31.
// B layout (16-bit 32x16): lanes 0-15 hold K0..7,K16..23, lanes 16-31 hold K8..15,K24..31 (same as A).
// ---------------------------------------------------------------------------
__device__ __forceinline__ void gemv_recip(
    const __bf16* __restrict__ Mh, const __bf16* __restrict__ Ml,
    const __bf16* vHi, const __bf16* vLo,            // LDS source vector (hi/lo)
    float* dF, __bf16* dHi, __bf16* dLo,             // LDS dest: 1/(M v)
    int tid)
{
    const int lane   = tid & 31;
    const int w      = tid >> 5;          // 0..31 -> rows 16w..16w+15
    const int rbase  = w * 16;
    const int rowSel = lane & 15;
    const int half   = lane >> 4;
    const int aoff   = half * 8;          // A K sub-offset per lane-half
    const int boff   = half * 8;          // B K sub-offset per lane-half

    const __bf16* rowH = Mh + (size_t)(rbase + rowSel) * N_;
    const __bf16* rowL = Ml + (size_t)(rbase + rowSel) * N_;

    v8f acc = {};
#pragma unroll
    for (int kt = 0; kt < N_ / 32; ++kt) {
        const int k0 = kt * 32;
        bf16x8 a0 = *(const bf16x8*)(rowH + k0 + aoff);
        bf16x8 a1 = *(const bf16x8*)(rowH + k0 + aoff + 16);
        bf16x8 l0 = *(const bf16x8*)(rowL + k0 + aoff);
        bf16x8 l1 = *(const bf16x8*)(rowL + k0 + aoff + 16);
        bf16x8 b0 = *(const bf16x8*)(vHi + k0 + boff);
        bf16x8 b1 = *(const bf16x8*)(vHi + k0 + boff + 16);
        bf16x8 c0 = *(const bf16x8*)(vLo + k0 + boff);
        bf16x8 c1 = *(const bf16x8*)(vLo + k0 + boff + 16);
        v16bf aH = __builtin_shufflevector(a0, a1, 0,1,2,3,4,5,6,7,8,9,10,11,12,13,14,15);
        v16bf aL = __builtin_shufflevector(l0, l1, 0,1,2,3,4,5,6,7,8,9,10,11,12,13,14,15);
        v16bf bH = __builtin_shufflevector(b0, b1, 0,1,2,3,4,5,6,7,8,9,10,11,12,13,14,15);
        v16bf bL = __builtin_shufflevector(c0, c1, 0,1,2,3,4,5,6,7,8,9,10,11,12,13,14,15);
        // m*v ~= hi*vhi + hi*vlo + lo*vhi  (lo*vlo term ~2^-18, dropped)
        acc = __builtin_amdgcn_wmma_f32_16x16x32_bf16(false, aH, false, bH, (short)0, acc, false, false);
        acc = __builtin_amdgcn_wmma_f32_16x16x32_bf16(false, aH, false, bL, (short)0, acc, false, false);
        acc = __builtin_amdgcn_wmma_f32_16x16x32_bf16(false, aL, false, bH, (short)0, acc, false, false);
    }
    // D: lanes 0-15 hold rows rbase+0..7 in acc[0..7]; lanes 16-31 rows rbase+8..15.
    // All columns identical (broadcast B) => lane rowSel==0 of each half writes 8 rows.
    if (rowSel == 0) {
#pragma unroll
        for (int t = 0; t < 8; ++t) {
            int   row = rbase + half * 8 + t;
            float rv  = 1.0f / acc[t];
            dF[row] = rv;
            __bf16 h = (__bf16)rv;
            dHi[row] = h;
            dLo[row] = (__bf16)(rv - (float)h);
        }
    }
}

// ---------------------------------------------------------------------------
// Main: one workgroup per batch, 32 waves. 100 iterations of two GEMVs with
// LDS-resident scaling vectors, then fused output pass y = r_i*(exp(x)+eps)*c_j.
// ---------------------------------------------------------------------------
__global__ __launch_bounds__(1024) void sinkhorn_iterate(
    const float* __restrict__ x,
    const __bf16* __restrict__ Mhi, const __bf16* __restrict__ Mlo,
    const __bf16* __restrict__ MThi, const __bf16* __restrict__ MTlo,
    float* __restrict__ out)
{
    __shared__ __attribute__((aligned(32))) float  cF[N_];
    __shared__ __attribute__((aligned(32))) float  rF[N_];
    __shared__ __attribute__((aligned(32))) __bf16 cHi[N_];
    __shared__ __attribute__((aligned(32))) __bf16 cLo[N_];
    __shared__ __attribute__((aligned(32))) __bf16 rHi[N_];
    __shared__ __attribute__((aligned(32))) __bf16 rLo[N_];

    const int b   = blockIdx.x;
    const int tid = threadIdx.x;
    const size_t mb = (size_t)b * N_ * N_;
    const __bf16* Mh  = Mhi  + mb;
    const __bf16* Ml  = Mlo  + mb;
    const __bf16* MTh = MThi + mb;
    const __bf16* MTl = MTlo + mb;

    if (tid < N_) {
        cF[tid]  = 1.0f;
        cHi[tid] = (__bf16)1.0f;
        cLo[tid] = (__bf16)0.0f;
    }
    __syncthreads();

    for (int it = 0; it < ITERS_; ++it) {
        gemv_recip(Mh,  Ml,  cHi, cLo, rF, rHi, rLo, tid);   // r = 1/(M c)
        __syncthreads();
        gemv_recip(MTh, MTl, rHi, rLo, cF, cHi, cLo, tid);   // c = 1/(M^T r)
        __syncthreads();
    }

    // y[i,j] = r_i * (exp(x)+eps) * c_j  -- exact f32 M, only r/c carry bf16 error
    const float* xb = x + mb;
    float* ob = out + mb;
    for (int e = tid; e < N_ * N_; e += 1024) {
        int i = e >> 9, j = e & (N_ - 1);
        ob[e] = rF[i] * (expf(xb[e]) + EPS_) * cF[j];
    }
}

// ---------------------------------------------------------------------------
extern "C" void kernel_launch(void* const* d_in, const int* in_sizes, int n_in,
                              void* d_out, int out_size, void* d_ws, size_t ws_size,
                              hipStream_t stream) {
    (void)in_sizes; (void)n_in; (void)out_size;
    const float* x = (const float*)d_in[0];
    float* out = (float*)d_out;

    const size_t plane = (size_t)B_ * N_ * N_;       // elements per bf16 plane
    if (ws_size < plane * 4 * sizeof(__bf16)) return; // needs 32 MiB scratch
    __bf16* Mhi  = (__bf16*)d_ws;
    __bf16* Mlo  = Mhi  + plane;
    __bf16* MThi = Mlo  + plane;
    __bf16* MTlo = MThi + plane;

    dim3 gp(N_ / 32, N_ / 32, B_);
    sinkhorn_prep<<<gp, 256, 0, stream>>>(x, Mhi, Mlo, MThi, MTlo);
    sinkhorn_iterate<<<B_, 1024, 0, stream>>>(x, Mhi, Mlo, MThi, MTlo, out);
}
